// NghSampler2_68178310857540
// MI455X (gfx1250) — matrix operand order repaired and checked by reading, not compiled, not run on hardware
//
#include <hip/hip_runtime.h>
#include <hip/hip_bf16.h>

// ---------------- problem constants (match setup_inputs) ----------------
#define BB      4
#define HH      512
#define WW      512
#define NN      2000
#define CCH     128
#define FH      128           // H/4
#define FW      128           // W/4
#define FSZ     (FH*FW)       // 16384 per channel
#define BN      (BB*NN)       // 8000
#define NPOS    29
#define NNEG    80
#define NCOLS   (1 + NNEG + BN)   // 8081
#define SC      ((size_t)BN * (size_t)NCOLS)

typedef __attribute__((ext_vector_type(16))) _Float16 v16h;
typedef __attribute__((ext_vector_type(8)))  _Float16 v8h;
typedef __attribute__((ext_vector_type(8)))  float    v8f;

// ---------------- workspace layout (byte offsets) ----------------
#define WS_POSOFF   0u          // int[NPOS*2]
#define WS_NEGOFF   4096u       // int[NNEG*2]
#define WS_VP1F     8192u       // float[BN*2]
#define WS_VP2F     73728u      // float[BN*2]
#define WS_QC0      139264u     // float[BN]
#define WS_VF0F     262144u     // float[BN*128]
#define WS_VF0H     4456448u    // _Float16[BN*128]
#define WS_DISTH    6553600u    // _Float16[BN*128]

// ============================================================
// Prep: build POS/NEG offset tables (single thread), vp1f/vp2f,
// and the border mask output.
// ============================================================
__global__ void k_prep(const int* __restrict__ pos1, int M,
                       int* __restrict__ posOff, int* __restrict__ negOff,
                       float* __restrict__ vp1f, float* __restrict__ vp2f,
                       float* __restrict__ mask_out)
{
    if (blockIdx.x == 0 && threadIdx.x == 0) {
        int np = 0, nn = 0;
        for (int j = -49; j <= 49; ++j) {
            for (int i = -49; i <= 49; ++i) {
                int d2 = i * i + j * j;
                if (d2 <= 9) {                       // POS_D^2 = 9
                    posOff[np * 2 + 0] = i;          // added to y (matches ref)
                    posOff[np * 2 + 1] = j;          // added to x
                    ++np;
                } else if (d2 >= 25 && d2 <= 49) {   // NEG_D^2=25 .. NGH^2=49
                    negOff[nn * 2 + 0] = i;
                    negOff[nn * 2 + 1] = j;
                    ++nn;
                }
            }
        }
    }
    for (int idx = blockIdx.x * blockDim.x + threadIdx.x; idx < BN;
         idx += gridDim.x * blockDim.x) {
        int b = idx / NN, n = idx % NN;
        int y1 = pos1[((size_t)b * M + n) * 2 + 0];
        int x1 = pos1[((size_t)b * M + n) * 2 + 1];
        vp1f[2 * idx + 0] = (float)y1;
        vp1f[2 * idx + 1] = (float)x1;
        int y2 = pos1[((size_t)b * M + (M - NN) + n) * 2 + 0];
        int x2 = pos1[((size_t)b * M + (M - NN) + n) * 2 + 1];
        vp2f[2 * idx + 0] = (float)y2;
        vp2f[2 * idx + 1] = (float)x2;
        mask_out[idx] = (x1 >= 0 && x1 < WW && y1 >= 0 && y1 < HH) ? 1.0f : 0.0f;
    }
}

// ============================================================
// gt fill: [BN, NCOLS] with column 0 = 1, rest 0 (as float)
// ============================================================
__global__ void k_gt(float* __restrict__ gt)
{
    size_t stride = (size_t)gridDim.x * blockDim.x;
    for (size_t i = (size_t)blockIdx.x * blockDim.x + threadIdx.x; i < SC; i += stride)
        gt[i] = (i % (size_t)NCOLS == 0) ? 1.0f : 0.0f;
}

// ---------------- bilinear helpers ----------------
__device__ inline void bil_setup(float fy, float fx,
                                 int& y0, int& x0, int& y1, int& x1,
                                 float& w00, float& w01, float& w10, float& w11)
{
    fy = fminf(fmaxf(fy, 0.0f), (float)(FH - 1));
    fx = fminf(fmaxf(fx, 0.0f), (float)(FW - 1));
    y0 = (int)floorf(fy); x0 = (int)floorf(fx);
    y1 = min(y0 + 1, FH - 1); x1 = min(x0 + 1, FW - 1);
    float wy = fy - (float)y0, wx = fx - (float)x0;
    w00 = (1.f - wy) * (1.f - wx); w01 = (1.f - wy) * wx;
    w10 = wy * (1.f - wx);         w11 = wy * wx;
}

__device__ inline float bil_tap(const float* __restrict__ f,
                                int y0, int x0, int y1, int x1,
                                float w00, float w01, float w10, float w11)
{
    return w00 * f[y0 * FW + x0] + w01 * f[y0 * FW + x1]
         + w10 * f[y1 * FW + x0] + w11 * f[y1 * FW + x1];
}

// ============================================================
// Sample + L2-normalize features at given points.
// One block (128 threads, thread = channel) per point.
// mode 0: feat0 @ pos0[:N]/4 -> vf0f (f32) + vf0h (f16) + qc0 (conf0)
// mode 1: feat1 @ pos1[M-N:]/4 -> dsth (f16) only
// ============================================================
__global__ void k_sample(const float* __restrict__ feat,
                         const float* __restrict__ conf,
                         const int*   __restrict__ pos, int M, int posBase,
                         float* __restrict__ outf, _Float16* __restrict__ outh,
                         float* __restrict__ qc0, int mode)
{
    int i = blockIdx.x;
    int b = i / NN, n = i % NN;
    int c = threadIdx.x;

    int py = pos[((size_t)b * M + posBase + n) * 2 + 0];
    int px = pos[((size_t)b * M + posBase + n) * 2 + 1];
    float fy = (float)py * 0.25f, fx = (float)px * 0.25f;

    int y0, x0, y1, x1; float w00, w01, w10, w11;
    bil_setup(fy, fx, y0, x0, y1, x1, w00, w01, w10, w11);

    const float* f = feat + ((size_t)b * CCH + c) * FSZ;
    float v = bil_tap(f, y0, x0, y1, x1, w00, w01, w10, w11);

    __shared__ float red[CCH];
    red[c] = v * v;
    __syncthreads();
    for (int s = CCH / 2; s > 0; s >>= 1) {
        if (c < s) red[c] += red[c + s];
        __syncthreads();
    }
    float nv = v / fmaxf(sqrtf(red[0]), 1e-12f);

    if (mode == 0) outf[(size_t)i * CCH + c] = nv;
    outh[(size_t)i * CCH + c] = (_Float16)nv;

    if (mode == 0 && c == 0) {
        const float* cf = conf + (size_t)b * FSZ;
        qc0[i] = bil_tap(cf, y0, x0, y1, x1, w00, w01, w10, w11);
    }
}

// ============================================================
// Candidate scoring: one wave (32 lanes) per point; each lane
// owns 4 channels. Writes pscore (col 0), nscores (cols 1..80),
// and qconf.
// ============================================================
__global__ void k_cand(const float* __restrict__ feat1,
                       const float* __restrict__ conf1,
                       const int*   __restrict__ posOff,
                       const int*   __restrict__ negOff,
                       const float* __restrict__ vp1f,
                       const float* __restrict__ vf0f,
                       const float* __restrict__ qc0,
                       float* __restrict__ scores,   // [BN, NCOLS]
                       float* __restrict__ qconf_out)
{
    int wave = threadIdx.x >> 5;
    int lane = threadIdx.x & 31;
    int i = blockIdx.x * 8 + wave;
    int b = i / NN;

    float vy = vp1f[2 * i + 0], vx = vp1f[2 * i + 1];
    const float4 vf4 = *(const float4*)(vf0f + (size_t)i * CCH + lane * 4);
    const float* fb = feat1 + ((size_t)b * CCH + lane * 4) * FSZ;

    float bestS = -1e30f;
    int bestP = 0;

    for (int p = 0; p < NPOS + NNEG; ++p) {
        int dy, dx;
        if (p < NPOS) { dy = posOff[2 * p];            dx = posOff[2 * p + 1]; }
        else          { dy = negOff[2 * (p - NPOS)];   dx = negOff[2 * (p - NPOS) + 1]; }
        float cy = fminf(fmaxf(vy + (float)dy, 0.0f), (float)(HH - 1));
        float cx = fminf(fmaxf(vx + (float)dx, 0.0f), (float)(WW - 1));
        int y0, x0, y1, x1; float w00, w01, w10, w11;
        bil_setup(cy * 0.25f, cx * 0.25f, y0, x0, y1, x1, w00, w01, w10, w11);

        float s2 = 0.f, dp = 0.f;
        #pragma unroll
        for (int q = 0; q < 4; ++q) {
            const float* f = fb + (size_t)q * FSZ;
            float v = bil_tap(f, y0, x0, y1, x1, w00, w01, w10, w11);
            s2 += v * v;
            float a = (q == 0) ? vf4.x : (q == 1) ? vf4.y : (q == 2) ? vf4.z : vf4.w;
            dp += a * v;
        }
        #pragma unroll
        for (int o = 16; o > 0; o >>= 1) {
            s2 += __shfl_xor(s2, o, 32);
            dp += __shfl_xor(dp, o, 32);
        }
        float sc = dp / fmaxf(sqrtf(s2), 1e-12f);

        if (p < NPOS) {
            if (sc > bestS) { bestS = sc; bestP = p; }
        } else if (lane == 0) {
            scores[(size_t)i * NCOLS + 1 + (p - NPOS)] = sc;
        }
    }

    if (lane == 0) {
        scores[(size_t)i * NCOLS + 0] = bestS;
        // qconf = (conf0@vp0 + conf1@sel)/2, sel = clamp(vp1 + posOff[bestP])
        float sy = fminf(fmaxf(vy + (float)posOff[2 * bestP + 0], 0.0f), (float)(HH - 1));
        float sx = fminf(fmaxf(vx + (float)posOff[2 * bestP + 1], 0.0f), (float)(WW - 1));
        int y0, x0, y1, x1; float w00, w01, w10, w11;
        bil_setup(sy * 0.25f, sx * 0.25f, y0, x0, y1, x1, w00, w01, w10, w11);
        const float* cf = conf1 + (size_t)b * FSZ;
        float c1 = bil_tap(cf, y0, x0, y1, x1, w00, w01, w10, w11);
        qconf_out[i] = 0.5f * (qc0[i] + c1);
    }
}

// ============================================================
// Distractor GEMM: dscores = vf0 @ dist.T  (8000x8000, K=128)
// fp16 inputs, f32 accum via v_wmma_f32_16x16x32_f16.
// 256 threads = 8 waves; wave tile 16x32 (TWO accumulators per
// wave, interleaved -> no WMMA->WMMA NOP bubble, A fragment
// reused across both tiles). Block tile 64x64.
// Fused spatial mask + scatter into scores cols 81..8080.
// ============================================================
__global__ void k_gemm(const _Float16* __restrict__ A,   // vf0h  [BN,128]
                       const _Float16* __restrict__ Bm,  // disth [BN,128]
                       const float* __restrict__ vp1f,
                       const float* __restrict__ vp2f,
                       float* __restrict__ scores)
{
    int wave = threadIdx.x >> 5;
    int lane = threadIdx.x & 31;
    int m16  = lane & 15;
    int kh   = lane >> 4;          // which half of the lane group

    int gi0 = blockIdx.y * 64 + (wave & 3) * 16;      // 4 wave-rows
    int gj0 = blockIdx.x * 64 + (wave >> 2) * 32;     // 2 wave-cols of 32

    // A: 16x32 f16 tile, contiguous k order (B matches) — lane holds row m16;
    //    elems 0..15 -> K = kh*16 + e
    const _Float16* ar  = A  + (size_t)(gi0 + m16) * CCH + kh * 16;
    // B: 32x16 f16, same contiguous k order as A — lanes 0-15 hold K=0..15, lanes 16-31 K=16..31
    const _Float16* br0 = Bm + (size_t)(gj0 + m16) * CCH + kh * 16;
    const _Float16* br1 = br0 + (size_t)16 * CCH;     // second column tile

    v8f acc0 = {};
    v8f acc1 = {};
    #pragma unroll
    for (int kk = 0; kk < CCH; kk += 32) {
        v8h alo = *(const v8h*)(ar + kk);
        v8h ahi = *(const v8h*)(ar + kk + 8);
        v8h b0lo = *(const v8h*)(br0 + kk);
        v8h b0hi = *(const v8h*)(br0 + kk + 8);
        v8h b1lo = *(const v8h*)(br1 + kk);
        v8h b1hi = *(const v8h*)(br1 + kk + 8);
        v16h a  = __builtin_shufflevector(alo,  ahi,  0,1,2,3,4,5,6,7,8,9,10,11,12,13,14,15);
        v16h b0 = __builtin_shufflevector(b0lo, b0hi, 0,1,2,3,4,5,6,7,8,9,10,11,12,13,14,15);
        v16h b1 = __builtin_shufflevector(b1lo, b1hi, 0,1,2,3,4,5,6,7,8,9,10,11,12,13,14,15);
        acc0 = __builtin_amdgcn_wmma_f32_16x16x32_f16(
            false, a, false, b0, (short)0, acc0, false, false);
        acc1 = __builtin_amdgcn_wmma_f32_16x16x32_f16(
            false, a, false, b1, (short)0, acc1, false, false);
    }

    // C layout: VGPR r, lanes 0-15 -> M=r ; lanes 16-31 -> M=r+8 ; N = lane&15
    int gja = gj0 + m16;
    int gjb = gja + 16;
    float y2a = vp2f[2 * gja + 0], x2a = vp2f[2 * gja + 1];
    float y2b = vp2f[2 * gjb + 0], x2b = vp2f[2 * gjb + 1];
    int bja = gja / NN;
    int bjb = gjb / NN;
    #pragma unroll
    for (int r = 0; r < 8; ++r) {
        int gi = gi0 + r + kh * 8;
        int bi = gi / NN;
        float y1 = vp1f[2 * gi + 0], x1 = vp1f[2 * gi + 1];
        float dxa = x2a - x1, dya = y2a - y1;
        float dxb = x2b - x1, dyb = y2b - y1;
        float d2a = dxa * dxa + dya * dya + ((bi != bja) ? 25.0f : 0.0f);
        float d2b = dxb * dxb + dyb * dyb + ((bi != bjb) ? 25.0f : 0.0f);
        float va = (d2a < 25.0f) ? 0.0f : acc0[r];
        float vb = (d2b < 25.0f) ? 0.0f : acc1[r];
        float* row = scores + (size_t)gi * NCOLS + 81;
        row[gja] = va;
        row[gjb] = vb;
    }
}

// ============================================================
extern "C" void kernel_launch(void* const* d_in, const int* in_sizes, int n_in,
                              void* d_out, int out_size, void* d_ws, size_t ws_size,
                              hipStream_t stream)
{
    const float* feat0 = (const float*)d_in[0];
    const float* feat1 = (const float*)d_in[1];
    const float* conf0 = (const float*)d_in[2];
    const float* conf1 = (const float*)d_in[3];
    const int*   pos0  = (const int*)d_in[4];
    const int*   pos1  = (const int*)d_in[5];
    int M = (n_in > 5 && in_sizes[5] > 0) ? in_sizes[5] / (BB * 2) : 2500;

    char* w = (char*)d_ws;
    int*      posOff = (int*)(w + WS_POSOFF);
    int*      negOff = (int*)(w + WS_NEGOFF);
    float*    vp1f   = (float*)(w + WS_VP1F);
    float*    vp2f   = (float*)(w + WS_VP2F);
    float*    qc0    = (float*)(w + WS_QC0);
    float*    vf0f   = (float*)(w + WS_VF0F);
    _Float16* vf0h   = (_Float16*)(w + WS_VF0H);
    _Float16* dsth   = (_Float16*)(w + WS_DISTH);

    float* scores    = (float*)d_out;                 // [BN, NCOLS]
    float* gt        = scores + SC;                   // [BN, NCOLS]
    float* mask_out  = gt + SC;                       // [BN]
    float* qconf_out = mask_out + BN;                 // [BN]

    k_prep<<<32, 256, 0, stream>>>(pos1, M, posOff, negOff, vp1f, vp2f, mask_out);
    k_gt<<<4096, 256, 0, stream>>>(gt);
    k_sample<<<BN, CCH, 0, stream>>>(feat0, conf0, pos0, M, 0,
                                     vf0f, vf0h, qc0, /*mode=*/0);
    k_sample<<<BN, CCH, 0, stream>>>(feat1, conf1, pos1, M, M - NN,
                                     vf0f, dsth, qc0, /*mode=*/1);
    k_cand<<<BN / 8, 256, 0, stream>>>(feat1, conf1, posOff, negOff,
                                       vp1f, vf0f, qc0, scores, qconf_out);
    dim3 gg(BN / 64, BN / 64);
    k_gemm<<<gg, 256, 0, stream>>>(vf0h, dsth, vp1f, vp2f, scores);
}
